// BlockForNormalWindow_87565793230982
// MI455X (gfx1250) — compile-verified, matrix operand order checked by reading, not run on hardware
//
#include <hip/hip_runtime.h>
#include <hip/hip_bf16.h>
#include <math.h>

typedef __attribute__((ext_vector_type(16))) __bf16 v16bf;
typedef __attribute__((ext_vector_type(8)))  float  v8f;

#define DIM   384
#define HEADS 6
#define HD    64
#define WIN   14
#define L     196
#define M_TOK 112896
#define ATT_SCALE 0.125f

// ---------------------------------------------------------------------------
// Fragment load helper: 16-bit WMMA operand layout (A 16x32 / B 32x16).
// lane m(or n) = lane&15 ; K half select kbase = 8*(lane>>4);
// element pairs: k = kbase + {0,2,4,6} and kbase + {16,18,20,22} as uints.
// rowptr must point at the first element of the row; k offsets are even ->
// every read is 4-byte aligned as long as the row start is 4-byte aligned.
// Works for both LDS and global pointers.
// ---------------------------------------------------------------------------
__device__ __forceinline__ v16bf frag_ld(const __bf16* rowptr, int kbase) {
  union { v16bf v; unsigned u[8]; } f;
#pragma unroll
  for (int p = 0; p < 8; ++p) {
    int k = kbase + ((p < 4) ? (2 * p) : (2 * p + 8));
    f.u[p] = *(const unsigned*)(rowptr + k);
  }
  return f.v;
}

__device__ __forceinline__ v8f wmma_bf16(v16bf a, v16bf b, v8f c) {
  return __builtin_amdgcn_wmma_f32_16x16x32_bf16(false, a, false, b,
                                                 (short)0, c, false, false);
}

// ---------------------------------------------------------------------------
// fp32 -> bf16 weight conversion
// ---------------------------------------------------------------------------
__global__ void cvt_kernel(const float* __restrict__ src,
                           __bf16* __restrict__ dst, int n) {
  int i = blockIdx.x * blockDim.x + threadIdx.x;
  if (i < n) dst[i] = (__bf16)src[i];
}

// ---------------------------------------------------------------------------
// LayerNorm over last dim (384), one wave per row, bf16 output.
// ---------------------------------------------------------------------------
__global__ __launch_bounds__(256) void ln_kernel(const float* __restrict__ x,
                                                 const float* __restrict__ w,
                                                 const float* __restrict__ b,
                                                 __bf16* __restrict__ out,
                                                 int M) {
  int row  = blockIdx.x * 8 + (threadIdx.x >> 5);
  int lane = threadIdx.x & 31;
  if (row >= M) return;
  const float* xr = x + (size_t)row * DIM;
  float v[12];
  float s = 0.f, ss = 0.f;
#pragma unroll
  for (int i = 0; i < 12; ++i) {
    float t = xr[lane + 32 * i];
    v[i] = t; s += t; ss += t * t;
  }
#pragma unroll
  for (int off = 16; off; off >>= 1) {
    s  += __shfl_xor(s,  off);
    ss += __shfl_xor(ss, off);
  }
  float mean = s * (1.0f / DIM);
  float var  = ss * (1.0f / DIM) - mean * mean;
  float rstd = rsqrtf(var + 1e-5f);
  __bf16* orow = out + (size_t)row * DIM;
#pragma unroll
  for (int i = 0; i < 12; ++i) {
    int c = lane + 32 * i;
    orow[c] = (__bf16)((v[i] - mean) * rstd * w[c] + b[c]);
  }
}

// ---------------------------------------------------------------------------
// bf16 GEMM: C[M,N] = A[M,K] * B[K,N] (+bias, +epilogue)
// block = 256 threads = 8 waves, tile 128x128, waves in 4(M)x2(N),
// each wave computes 32x64 via 2x4 v_wmma_f32_16x16x32_bf16 tiles.
// EPI: 0 = bias -> bf16 ; 1 = bias + residual -> fp32 ; 2 = bias+GELU -> bf16
// ---------------------------------------------------------------------------
template <int EPI>
__global__ __launch_bounds__(256) void gemm_kernel(
    const __bf16* __restrict__ A, const __bf16* __restrict__ B, int N, int K,
    const float* __restrict__ bias, const float* __restrict__ resid,
    float* __restrict__ outF, __bf16* __restrict__ outB) {
  __shared__ __bf16 As[128][40];  // A tile, row-major [m][k]
  __shared__ __bf16 Bs[128][40];  // B tile transposed [n][k]

  const int tid  = threadIdx.x;
  const int lane = tid & 31;
  const int wv   = tid >> 5;
  const int wm   = wv >> 1, wn = wv & 1;
  const int m0   = blockIdx.x * 128;
  const int n0   = blockIdx.y * 128;
  const int mm   = lane & 15;
  const int kb8  = (lane >> 4) * 8;

  v8f acc[2][4];
#pragma unroll
  for (int i = 0; i < 2; ++i)
#pragma unroll
    for (int j = 0; j < 4; ++j)
      acc[i][j] = (v8f){0.f, 0.f, 0.f, 0.f, 0.f, 0.f, 0.f, 0.f};

  for (int k0 = 0; k0 < K; k0 += 32) {
    // stage A: 128 rows x 32 k (2048 uints)
#pragma unroll
    for (int p = 0; p < 8; ++p) {
      int idx = p * 256 + tid;
      int r = idx >> 4, kw = (idx & 15) * 2;
      unsigned u = *(const unsigned*)(A + (size_t)(m0 + r) * K + k0 + kw);
      *(unsigned*)(&As[r][kw]) = u;
    }
    // stage B transposed: 32 k-rows x 128 n
#pragma unroll
    for (int p = 0; p < 8; ++p) {
      int idx = p * 256 + tid;
      int kr = idx >> 6, nn = (idx & 63) * 2;
      union { unsigned u; __bf16 h[2]; } cv;
      cv.u = *(const unsigned*)(B + (size_t)(k0 + kr) * N + n0 + nn);
      Bs[nn][kr]     = cv.h[0];
      Bs[nn + 1][kr] = cv.h[1];
    }
    __syncthreads();

    v16bf af[2], bfr[4];
#pragma unroll
    for (int i = 0; i < 2; ++i)
      af[i] = frag_ld(&As[wm * 32 + i * 16 + mm][0], kb8);
#pragma unroll
    for (int j = 0; j < 4; ++j)
      bfr[j] = frag_ld(&Bs[wn * 64 + j * 16 + mm][0], kb8);
#pragma unroll
    for (int i = 0; i < 2; ++i)
#pragma unroll
      for (int j = 0; j < 4; ++j)
        acc[i][j] = wmma_bf16(af[i], bfr[j], acc[i][j]);
    __syncthreads();
  }

  // epilogue: D layout -> lane col = lane&15, VGPR e -> row e + 8*(lane>>4)
#pragma unroll
  for (int i = 0; i < 2; ++i)
#pragma unroll
    for (int j = 0; j < 4; ++j)
#pragma unroll
      for (int e = 0; e < 8; ++e) {
        int row = m0 + wm * 32 + i * 16 + e + ((lane >> 4) * 8);
        int col = n0 + wn * 64 + j * 16 + mm;
        float v = acc[i][j][e] + bias[col];
        if (EPI == 2) v = 0.5f * v * (1.0f + erff(v * 0.70710678118f));
        if (EPI == 1) {
          v += resid[(size_t)row * N + col];
          outF[(size_t)row * N + col] = v;
        } else {
          outB[(size_t)row * N + col] = (__bf16)v;
        }
      }
}

// ---------------------------------------------------------------------------
// Windowed attention with decomposed rel-pos, flash-style online softmax.
// One block per (window, head); 13 waves; each wave owns a 16-query chunk.
// q and k fragments are pair-loaded straight from global (d-contiguous, L0/L2
// hot); only V (needs key-contiguous pairs -> transposed) plus the rel-pos
// tables and the per-wave P scratch live in LDS: 65,024 B total (<= 64 KiB).
// keys padded to 224 = 7 chunks of 32; masked keys get -1e30 before softmax.
// ---------------------------------------------------------------------------
__global__ __launch_bounds__(416) void attn_kernel(
    const __bf16* __restrict__ qkv, const float* __restrict__ relPH,
    const float* __restrict__ relPW, __bf16* __restrict__ out) {
  extern __shared__ char smem[];
  __bf16* vT_s = (__bf16*)(smem);            // [64][226]   28928 B
  float*  rh_s = (float*)(smem + 28928);     // [196][14]   10976 B
  float*  rw_s = (float*)(smem + 39904);     // [196][14]   10976 B
  __bf16* p_s  = (__bf16*)(smem + 50880);    // [13][16][34] 14144 B

  const int tid  = threadIdx.x;
  const int win  = blockIdx.x;
  const int head = blockIdx.y;
  const int wy = win / 24, wx = win % 24;
  const size_t hoff = (size_t)head * HD;

  // stage v transposed [d][key]
  for (int idx = tid; idx < 224 * 64; idx += 416) {
    int r = idx >> 6, d = idx & 63;
    __bf16 val = (__bf16)0.f;
    if (r < L) {
      int i = r / WIN, j = r % WIN;
      size_t tok = (size_t)((wy * WIN + i) * 336 + wx * WIN + j);
      val = qkv[tok * 1152 + 768 + hoff + d];
    }
    vT_s[d * 226 + r] = val;
  }

  // rel-pos bias tables: rh_s[q][ki] = dot(q_vec, rel_pos_h[i-ki+13])
  // (q read straight from global; unscaled q, matching the reference)
  for (int p = tid; p < L * WIN; p += 416) {
    int ql = p / WIN, kk = p % WIN;
    int i = ql / WIN, j = ql % WIN;
    size_t tok = (size_t)((wy * WIN + i) * 336 + wx * WIN + j);
    const __bf16* qg = qkv + tok * 1152 + hoff;
    const float* rh = relPH + (size_t)(i - kk + WIN - 1) * HD;
    const float* rw = relPW + (size_t)(j - kk + WIN - 1) * HD;
    float sh = 0.f, sw = 0.f;
#pragma unroll
    for (int d = 0; d < HD; ++d) {
      float qd = (float)qg[d];
      sh += qd * rh[d];
      sw += qd * rw[d];
    }
    rh_s[p] = sh;
    rw_s[p] = sw;
  }
  __syncthreads();

  const int wv = tid >> 5, lane = tid & 31;
  const int mm = lane & 15, kb8 = (lane >> 4) * 8;
  const int half8 = (lane >> 4) * 8;

  // q A-fragments straight from global (row clamped; bad rows are masked out)
  v16bf qa0, qa1;
  {
    int qrow = wv * 16 + mm;
    if (qrow > L - 1) qrow = L - 1;
    int i = qrow / WIN, j = qrow % WIN;
    size_t tok = (size_t)((wy * WIN + i) * 336 + wx * WIN + j);
    const __bf16* qg = qkv + tok * 1152 + hoff;
    qa0 = frag_ld(qg, kb8);
    qa1 = frag_ld(qg + 32, kb8);
  }

  v8f o0 = (v8f){0.f,0.f,0.f,0.f,0.f,0.f,0.f,0.f};
  v8f o1 = o0, o2 = o0, o3 = o0;
  float mI[8], lI[8];
#pragma unroll
  for (int e = 0; e < 8; ++e) { mI[e] = -1e30f; lI[e] = 0.f; }

  for (int kc = 0; kc < 7; ++kc) {
    const int kb = kc * 32;
    // scores: two 16x16 tiles, each = qa0*K(d0..31) + qa1*K(d32..63)
    // k B-fragments pair-load along d straight from global (clamped key).
    v8f s0, s1;
#pragma unroll
    for (int t = 0; t < 2; ++t) {
      int kcol = kb + t * 16 + mm;
      if (kcol > L - 1) kcol = L - 1;
      int i = kcol / WIN, j = kcol % WIN;
      size_t tok = (size_t)((wy * WIN + i) * 336 + wx * WIN + j);
      const __bf16* kg = qkv + tok * 1152 + 384 + hoff;
      v8f c = (v8f){0.f,0.f,0.f,0.f,0.f,0.f,0.f,0.f};
      c = wmma_bf16(qa0, frag_ld(kg, kb8), c);
      c = wmma_bf16(qa1, frag_ld(kg + 32, kb8), c);
      if (t == 0) s0 = c; else s1 = c;
    }
    // online softmax update (rows live across 16-lane halves)
#pragma unroll
    for (int e = 0; e < 8; ++e) {
      int row = wv * 16 + e + half8;
      int key0 = kb + mm, key1 = kb + 16 + mm;
      float v0 = -1e30f, v1 = -1e30f;
      if (row < L) {
        if (key0 < L) {
          int ki = key0 / WIN, kj = key0 % WIN;
          v0 = s0[e] * ATT_SCALE + rh_s[row * WIN + ki] + rw_s[row * WIN + kj];
        }
        if (key1 < L) {
          int ki = key1 / WIN, kj = key1 % WIN;
          v1 = s1[e] * ATT_SCALE + rh_s[row * WIN + ki] + rw_s[row * WIN + kj];
        }
      }
      float rm = fmaxf(v0, v1);
      rm = fmaxf(rm, __shfl_xor(rm, 1));
      rm = fmaxf(rm, __shfl_xor(rm, 2));
      rm = fmaxf(rm, __shfl_xor(rm, 4));
      rm = fmaxf(rm, __shfl_xor(rm, 8));
      float newm = fmaxf(mI[e], rm);
      float scl = __expf(mI[e] - newm);
      float p0 = __expf(v0 - newm);
      float p1 = __expf(v1 - newm);
      float rs = p0 + p1;
      rs += __shfl_xor(rs, 1);
      rs += __shfl_xor(rs, 2);
      rs += __shfl_xor(rs, 4);
      rs += __shfl_xor(rs, 8);
      lI[e] = lI[e] * scl + rs;
      mI[e] = newm;
      o0[e] *= scl; o1[e] *= scl; o2[e] *= scl; o3[e] *= scl;
      // dump P (D-layout) to per-wave LDS scratch for A-fragment reload
      __bf16* pr = p_s + (wv * 16 + e + half8) * 34;
      pr[mm]      = (__bf16)p0;
      pr[16 + mm] = (__bf16)p1;
    }
    // P (16x32) * V (32x64): same-wave LDS ops are in order, no barrier
    v16bf pa = frag_ld(p_s + (wv * 16 + mm) * 34, kb8);
    o0 = wmma_bf16(pa, frag_ld(vT_s + (0 * 16 + mm) * 226 + kb, kb8), o0);
    o1 = wmma_bf16(pa, frag_ld(vT_s + (1 * 16 + mm) * 226 + kb, kb8), o1);
    o2 = wmma_bf16(pa, frag_ld(vT_s + (2 * 16 + mm) * 226 + kb, kb8), o2);
    o3 = wmma_bf16(pa, frag_ld(vT_s + (3 * 16 + mm) * 226 + kb, kb8), o3);
  }

  // normalize and scatter back to token-major [tok][head*64+d] bf16
#pragma unroll
  for (int e = 0; e < 8; ++e) {
    int row = wv * 16 + e + half8;
    if (row < L) {
      int i = row / WIN, j = row % WIN;
      size_t tok = (size_t)((wy * WIN + i) * 336 + wx * WIN + j);
      float inv = 1.0f / lI[e];
      __bf16* op = out + tok * DIM + hoff;
      op[0 * 16 + mm] = (__bf16)(o0[e] * inv);
      op[1 * 16 + mm] = (__bf16)(o1[e] * inv);
      op[2 * 16 + mm] = (__bf16)(o2[e] * inv);
      op[3 * 16 + mm] = (__bf16)(o3[e] * inv);
    }
  }
}

// ---------------------------------------------------------------------------
extern "C" void kernel_launch(void* const* d_in, const int* in_sizes, int n_in,
                              void* d_out, int out_size, void* d_ws,
                              size_t ws_size, hipStream_t stream) {
  const float* x      = (const float*)d_in[0];
  const float* qkv_w  = (const float*)d_in[1];
  const float* qkv_b  = (const float*)d_in[2];
  const float* proj_w = (const float*)d_in[3];
  const float* proj_b = (const float*)d_in[4];
  const float* rel_h  = (const float*)d_in[5];
  const float* rel_w  = (const float*)d_in[6];
  const float* n1w    = (const float*)d_in[7];
  const float* n1b    = (const float*)d_in[8];
  const float* n2w    = (const float*)d_in[9];
  const float* n2b    = (const float*)d_in[10];
  const float* fc1_w  = (const float*)d_in[11];
  const float* fc1_b  = (const float*)d_in[12];
  const float* fc2_w  = (const float*)d_in[13];
  const float* fc2_b  = (const float*)d_in[14];

  char* ws = (char*)d_ws;
  __bf16* wq   = (__bf16*)(ws + 0);          // 384*1152 bf16
  __bf16* wp   = (__bf16*)(ws + 884736);     // 384*384
  __bf16* w1   = (__bf16*)(ws + 1179648);    // 384*1536
  __bf16* w2   = (__bf16*)(ws + 2359296);    // 1536*384
  __bf16* bufA = (__bf16*)(ws + 3538944);    // 112896*384 bf16 (xn1/attn/xn2)
  float*  x2   = (float*)(ws + 90243072);    // 112896*384 fp32
  __bf16* bufB = (__bf16*)(ws + 263651328);  // qkv (112896*1152) / h1 (112896*1536)

  // weights fp32 -> bf16
  cvt_kernel<<<(442368 + 255) / 256, 256, 0, stream>>>(qkv_w, wq, 442368);
  cvt_kernel<<<(147456 + 255) / 256, 256, 0, stream>>>(proj_w, wp, 147456);
  cvt_kernel<<<(589824 + 255) / 256, 256, 0, stream>>>(fc1_w, w1, 589824);
  cvt_kernel<<<(589824 + 255) / 256, 256, 0, stream>>>(fc2_w, w2, 589824);

  // LN1 -> bf16
  ln_kernel<<<M_TOK / 8, 256, 0, stream>>>(x, n1w, n1b, bufA, M_TOK);

  // QKV: [112896,384] x [384,1152] + bias -> bf16
  gemm_kernel<0><<<dim3(882, 9), 256, 0, stream>>>(bufA, wq, 1152, 384, qkv_b,
                                                   nullptr, nullptr, bufB);

  // windowed attention with rel-pos -> bf16 token-major
  attn_kernel<<<dim3(576, 6), 416, 65024, stream>>>(bufB, rel_h, rel_w, bufA);

  // proj + residual(x) -> fp32 x2
  gemm_kernel<1><<<dim3(882, 3), 256, 0, stream>>>(bufA, wp, 384, 384, proj_b,
                                                   x, x2, nullptr);

  // LN2 -> bf16
  ln_kernel<<<M_TOK / 8, 256, 0, stream>>>(x2, n2w, n2b, bufA, M_TOK);

  // fc1 + GELU -> bf16
  gemm_kernel<2><<<dim3(882, 12), 256, 0, stream>>>(bufA, w1, 1536, 384, fc1_b,
                                                    nullptr, nullptr, bufB);

  // fc2 + residual(x2) -> fp32 output
  gemm_kernel<1><<<dim3(882, 3), 256, 0, stream>>>(bufB, w2, 384, 1536, fc2_b,
                                                   x2, (float*)d_out, nullptr);
}
